// DGCNNPartSeg_12317966205321
// MI455X (gfx1250) — compile-verified, matrix operand order checked by reading, not run on hardware
//
#include <hip/hip_runtime.h>
#include <hip/hip_bf16.h>

// ---------------------------------------------------------------------------
// DGCNN part-seg forward for MI455X (gfx1250).
// All channel-mixing 1x1 convs are WMMA GEMMs (v_wmma_f32_16x16x32_f16):
//   activations channels-last f16, K-dim padded to multiples of 32.
// Main GEMM uses 2x4 register blocking: one wave computes 2 M-tiles x 4
// N-tiles -> 8 WMMAs per k-step against 6 fragment loads (2 A + 4 B),
// 4x less A traffic / 2x less B traffic than the 1-tile version.
// Training-mode BN: GEMM -> f32 scratch -> deterministic per-channel
// reduction -> fused BN+ReLU+f16 repack (handles concat offset + zero pad).
//
// Assumed flat input order (setup_inputs insertion order, params flattened
// in order): d_in[0]=points(16,3,2048) f32, d_in[1]=cls_label(16) i32, then:
//  [0..2]tnet_edge1(64,6) [3..5]tnet_edge2(128,64) [6..8]tnet_local(1024,128)
//  [9..11]tnet_g1(512,1024) [12..14]tnet_g2(256,512) [15,16]tnet_linear(9,256)
//  [17..19]blk1a(64,6) [20..22]blk1b(64,64) [23..25]blk2a(64,128)
//  [26..28]blk2b(64,64) [29..31]blk3(64,128) [32..34]label(64,16)
//  [35..37]local(1024,192) [38..40]seg1(256,1280) [41..43]seg2(256,256)
//  [44..46]conv_seg(128,256) [47,48]logit(50,128)+b(50)
// Workspace ~0.9 GB (two 168MB f16 edge buffers + 336MB f32 GEMM scratch).
// ---------------------------------------------------------------------------

typedef _Float16 h16;
typedef __attribute__((ext_vector_type(16))) _Float16 v16h;
typedef __attribute__((ext_vector_type(8)))  _Float16 v8h;
typedef __attribute__((ext_vector_type(8)))  float    v8f;

#define BD 16
#define NPTS 2048
#define KNN 20
#define NSEG 50
#define NCLS 16
#define BN_PTS (BD * NPTS)        /* 32768  */
#define M_EDGE (BD * NPTS * KNN)  /* 655360 */
#define BN_EPS 1e-5f

static __device__ __forceinline__ v16h ld_frag(const h16* p) {
  v8h lo = *(const v8h*)(p);
  v8h hi = *(const v8h*)(p + 16);
  v16h f;
#pragma unroll
  for (int j = 0; j < 8; ++j) { f[j] = lo[j]; f[j + 8] = hi[j]; }
  return f;
}

// ------------- GEMM 2x4: Y[M,Copad] = X[M,Kp] * W[Copad,Kp]^T --------------
// One wave per (2 M-tiles x 4 N-tiles). Requires M%32==0, Copad%64==0,
// Kp%32==0, buffers 16B aligned. 8 WMMAs per k-step.
__global__ __launch_bounds__(32) void gemm_wmma_2x4(
    const h16* __restrict__ X, const h16* __restrict__ Wt,
    float* __restrict__ Y, int Kp, int Copad) {
  const int lane = threadIdx.x & 31;
  const int tm = blockIdx.x, tn = blockIdx.y;
  const int r = lane & 15, hf = lane >> 4;
  const h16* a0 = X + (size_t)(tm * 32 + r) * Kp + 8 * hf;
  const h16* a1 = X + (size_t)(tm * 32 + 16 + r) * Kp + 8 * hf;
  const h16* b0 = Wt + (size_t)(tn * 64 + r) * Kp + 8 * hf;
  const h16* b1 = Wt + (size_t)(tn * 64 + 16 + r) * Kp + 8 * hf;
  const h16* b2 = Wt + (size_t)(tn * 64 + 32 + r) * Kp + 8 * hf;
  const h16* b3 = Wt + (size_t)(tn * 64 + 48 + r) * Kp + 8 * hf;
  v8f acc0[4] = {}, acc1[4] = {};
  for (int k0 = 0; k0 < Kp; k0 += 32) {
    if (k0 + 32 < Kp) {
      __builtin_prefetch(a0 + k0 + 32, 0, 1);
      __builtin_prefetch(a1 + k0 + 32, 0, 1);
    }
    v16h fa0 = ld_frag(a0 + k0);
    v16h fa1 = ld_frag(a1 + k0);
    v16h fb0 = ld_frag(b0 + k0);
    v16h fb1 = ld_frag(b1 + k0);
    v16h fb2 = ld_frag(b2 + k0);
    v16h fb3 = ld_frag(b3 + k0);
    acc0[0] = __builtin_amdgcn_wmma_f32_16x16x32_f16(false, fa0, false, fb0, (short)0, acc0[0], false, false);
    acc1[0] = __builtin_amdgcn_wmma_f32_16x16x32_f16(false, fa1, false, fb0, (short)0, acc1[0], false, false);
    acc0[1] = __builtin_amdgcn_wmma_f32_16x16x32_f16(false, fa0, false, fb1, (short)0, acc0[1], false, false);
    acc1[1] = __builtin_amdgcn_wmma_f32_16x16x32_f16(false, fa1, false, fb1, (short)0, acc1[1], false, false);
    acc0[2] = __builtin_amdgcn_wmma_f32_16x16x32_f16(false, fa0, false, fb2, (short)0, acc0[2], false, false);
    acc1[2] = __builtin_amdgcn_wmma_f32_16x16x32_f16(false, fa1, false, fb2, (short)0, acc1[2], false, false);
    acc0[3] = __builtin_amdgcn_wmma_f32_16x16x32_f16(false, fa0, false, fb3, (short)0, acc0[3], false, false);
    acc1[3] = __builtin_amdgcn_wmma_f32_16x16x32_f16(false, fa1, false, fb3, (short)0, acc1[3], false, false);
  }
#pragma unroll
  for (int j = 0; j < 4; ++j) {
    float* y0 = Y + (size_t)(tm * 32 + 8 * hf) * Copad + tn * 64 + j * 16 + r;
    float* y1 = Y + (size_t)(tm * 32 + 16 + 8 * hf) * Copad + tn * 64 + j * 16 + r;
#pragma unroll
    for (int v = 0; v < 8; ++v) {
      y0[(size_t)v * Copad] = acc0[j][v];
      y1[(size_t)v * Copad] = acc1[j][v];
    }
  }
}

// ---------------- GEMM 1x1 fallback (small M or narrow Copad) --------------
__global__ __launch_bounds__(32) void gemm_wmma_1(
    const h16* __restrict__ X, const h16* __restrict__ Wt,
    float* __restrict__ Y, int Kp, int Copad) {
  const int lane = threadIdx.x & 31;
  const int tm = blockIdx.x, tn = blockIdx.y;
  const int r = lane & 15, hf = lane >> 4;
  const h16* ax = X  + (size_t)(tm * 16 + r) * Kp + 8 * hf;
  const h16* bx = Wt + (size_t)(tn * 16 + r) * Kp + 8 * hf;
  v8f acc = {};
  for (int k0 = 0; k0 < Kp; k0 += 32) {
    v16h a = ld_frag(ax + k0);
    v16h b = ld_frag(bx + k0);
    acc = __builtin_amdgcn_wmma_f32_16x16x32_f16(
        false, a, false, b, (short)0, acc, false, false);
  }
  float* yp = Y + (size_t)(tm * 16 + 8 * hf) * Copad + tn * 16 + r;
#pragma unroll
  for (int v = 0; v < 8; ++v) yp[(size_t)v * Copad] = acc[v];
}

// -------- deterministic per-channel mean / rsqrt(var+eps) over M rows ------
__global__ __launch_bounds__(256) void col_stats(
    const float* __restrict__ Y, int M, int stride,
    float* __restrict__ mean, float* __restrict__ rstd) {
  __shared__ float sh[256], sh2[256];
  const int c = blockIdx.x;
  float s = 0.f, s2 = 0.f;
  for (int m = threadIdx.x; m < M; m += 256) {
    float v = Y[(size_t)m * stride + c];
    s += v; s2 += v * v;
  }
  sh[threadIdx.x] = s; sh2[threadIdx.x] = s2;
  __syncthreads();
  for (int o = 128; o > 0; o >>= 1) {
    if ((int)threadIdx.x < o) {
      sh[threadIdx.x]  += sh[threadIdx.x + o];
      sh2[threadIdx.x] += sh2[threadIdx.x + o];
    }
    __syncthreads();
  }
  if (threadIdx.x == 0) {
    float mu  = sh[0] / (float)M;
    float var = sh2[0] / (float)M - mu * mu;
    mean[c] = mu;
    rstd[c] = rsqrtf(var + BN_EPS);
  }
}

// ------ BN + ReLU + repack f32 -> f16 (dst col offset, zero pad width) -----
__global__ void bn_relu_pack(
    const float* __restrict__ Y, int srcStride, int Co,
    const float* __restrict__ gam, const float* __restrict__ bet,
    const float* __restrict__ mean, const float* __restrict__ rstd,
    h16* __restrict__ Xo, int dstStride, int dstOff, int padTo, int Mrows) {
  long long t = (long long)blockIdx.x * blockDim.x + threadIdx.x;
  long long total = (long long)Mrows * padTo;
  if (t >= total) return;
  int c = (int)(t % padTo);
  long long m = t / padTo;
  float v = 0.f;
  if (c < Co) {
    float y = Y[m * srcStride + c];
    y = gam[c] * (y - mean[c]) * rstd[c] + bet[c];
    v = y > 0.f ? y : 0.f;
  }
  Xo[m * dstStride + dstOff + c] = (h16)v;
}

// ---------------- weight pack: f32 (O,I) -> f16 (Opad,Ipad) ----------------
__global__ void pack_w(const float* __restrict__ W, h16* __restrict__ Wh,
                       int O, int I, int Opad, int Ipad) {
  long long t = (long long)blockIdx.x * blockDim.x + threadIdx.x;
  long long tot = (long long)Opad * Ipad;
  if (t >= tot) return;
  int i = (int)(t % Ipad), o = (int)(t / Ipad);
  float v = (o < O && i < I) ? W[(size_t)o * I + i] : 0.f;
  Wh[t] = (h16)v;
}

// ------------------ (B,3,N) f32 -> channels-last (B*N,3) -------------------
__global__ void transpose_in(const float* __restrict__ pts,
                             float* __restrict__ xt) {
  int t = blockIdx.x * blockDim.x + threadIdx.x;
  if (t >= BN_PTS * 3) return;
  int c = t % 3, bn = t / 3;
  int b = bn / NPTS, n = bn % NPTS;
  xt[t] = pts[((size_t)b * 3 + c) * NPTS + n];
}

// -------------------- kNN (top-KNN incl. self, per point) ------------------
__global__ void knn_kernel(const float* __restrict__ xt, int Cc, int xstride,
                           int* __restrict__ idx) {
  int n = blockIdx.x * blockDim.x + threadIdx.x;
  int b = blockIdx.y;
  if (n >= NPTS) return;
  float xq[64];
  const float* xp = xt + ((size_t)b * NPTS + n) * xstride;
  for (int c = 0; c < Cc; ++c) xq[c] = xp[c];
  float bd[KNN]; int bi[KNN];
  for (int k = 0; k < KNN; ++k) { bd[k] = 3.0e38f; bi[k] = 0; }
  const float* xb = xt + (size_t)b * NPTS * xstride;
  for (int m = 0; m < NPTS; ++m) {
    const float* xm = xb + (size_t)m * xstride;
    float d = 0.f;
    for (int c = 0; c < Cc; ++c) { float dv = xq[c] - xm[c]; d += dv * dv; }
    if (d < bd[KNN - 1]) {
      int pos = KNN - 1;
      while (pos > 0 && bd[pos - 1] > d) {
        bd[pos] = bd[pos - 1]; bi[pos] = bi[pos - 1]; --pos;
      }
      bd[pos] = d; bi[pos] = m;
    }
  }
  int* op = idx + ((size_t)b * NPTS + n) * KNN;
  for (int k = 0; k < KNN; ++k) op[k] = bi[k];
}

// -------- edge features concat[central, nbr-central] -> f16, zero-pad ------
__global__ void build_edge(const float* __restrict__ xt, int Cc, int xstride,
                           const int* __restrict__ idx,
                           h16* __restrict__ Xo, int Kp) {
  long long t = (long long)blockIdx.x * blockDim.x + threadIdx.x;
  if (t >= (long long)M_EDGE) return;
  long long bn = t / KNN;
  int b = (int)(bn / NPTS);
  int nb = idx[t];
  const float* xc = xt + bn * xstride;
  const float* xn = xt + ((size_t)b * NPTS + nb) * xstride;
  h16* o = Xo + t * Kp;
  for (int c = 0; c < Cc; ++c) {
    o[c]      = (h16)xc[c];
    o[Cc + c] = (h16)(xn[c] - xc[c]);
  }
  for (int c = 2 * Cc; c < Kp; ++c) o[c] = (h16)0.f;
}

// ----------------- max over KNN -> f16 (+ optional f32 copy) ---------------
__global__ void max_over_k(const h16* __restrict__ Xi, int srcStride, int cols,
                           h16* __restrict__ Xo, int dstStride, int dstOff,
                           float* __restrict__ xf32, int f32Stride) {
  int bn = blockIdx.x;
  for (int c = threadIdx.x; c < cols; c += blockDim.x) {
    float mx = -3.0e38f;
    const h16* src = Xi + (size_t)bn * KNN * srcStride + c;
    for (int k = 0; k < KNN; ++k) mx = fmaxf(mx, (float)src[(size_t)k * srcStride]);
    Xo[(size_t)bn * dstStride + dstOff + c] = (h16)mx;
    if (xf32) xf32[(size_t)bn * f32Stride + c] = mx;
  }
}

// ------------------------- max over N per batch ----------------------------
__global__ void max_over_n(const h16* __restrict__ Xi, int stride, int cols,
                           h16* __restrict__ out, int ostride) {
  int c = blockIdx.x * blockDim.x + threadIdx.x;
  int b = blockIdx.y;
  if (c >= cols) return;
  float mx = -3.0e38f;
  const h16* src = Xi + (size_t)b * NPTS * stride + c;
  for (int n = 0; n < NPTS; ++n) mx = fmaxf(mx, (float)src[(size_t)n * stride]);
  out[(size_t)b * ostride + c] = (h16)mx;
}

// --------------- trans = Y@Wl^T + bl, reshaped 3x3, +Identity --------------
__global__ void make_trans_k(const float* __restrict__ Y,
                             const float* __restrict__ bl,
                             float* __restrict__ tr) {
  int t = threadIdx.x;
  if (t >= BD * 9) return;
  int b = t / 9, e = t % 9;
  float v = Y[(size_t)b * 16 + e] + bl[e];
  if (e == 0 || e == 4 || e == 8) v += 1.f;
  tr[t] = v;
}

__global__ void apply_trans_k(const float* __restrict__ xi,
                              const float* __restrict__ tr,
                              float* __restrict__ xo) {
  int t = blockIdx.x * blockDim.x + threadIdx.x;
  if (t >= BN_PTS) return;
  int b = t / NPTS;
  const float* m = tr + b * 9;
  float x0 = xi[(size_t)t * 3], x1 = xi[(size_t)t * 3 + 1], x2 = xi[(size_t)t * 3 + 2];
  xo[(size_t)t * 3]     = m[0] * x0 + m[1] * x1 + m[2] * x2;
  xo[(size_t)t * 3 + 1] = m[3] * x0 + m[4] * x1 + m[5] * x2;
  xo[(size_t)t * 3 + 2] = m[6] * x0 + m[7] * x1 + m[8] * x2;
}

__global__ void onehot_k(const int* __restrict__ cls, h16* __restrict__ X) {
  int t = blockIdx.x * blockDim.x + threadIdx.x;
  if (t >= BD * 32) return;
  int c = t % 32, b = t / 32;
  X[t] = (h16)((c < NCLS && cls[b] == c) ? 1.f : 0.f);
}

// ------------- xc = concat[gmax(1024), lab(64), cat(192)] per point --------
__global__ void build_xc_k(const h16* __restrict__ gm, const h16* __restrict__ lab,
                           const h16* __restrict__ cat, h16* __restrict__ xc) {
  long long t = (long long)blockIdx.x * blockDim.x + threadIdx.x;
  long long tot = (long long)BN_PTS * 1280;
  if (t >= tot) return;
  int c = (int)(t % 1280);
  long long bn = t / 1280;
  int b = (int)(bn / NPTS);
  h16 v;
  if (c < 1024)      v = gm[b * 1024 + c];
  else if (c < 1088) v = lab[b * 64 + (c - 1024)];
  else               v = cat[bn * 192 + (c - 1088)];
  xc[t] = v;
}

// --------- add bias, transpose (B*N,64)->(B,50,N), write final out ---------
__global__ void finalize_k(const float* __restrict__ Y,
                           const float* __restrict__ bo,
                           float* __restrict__ out) {
  long long t = (long long)blockIdx.x * blockDim.x + threadIdx.x;
  long long tot = (long long)BD * NSEG * NPTS;
  if (t >= tot) return;
  int n = (int)(t % NPTS);
  int o = (int)((t / NPTS) % NSEG);
  int b = (int)(t / ((long long)NSEG * NPTS));
  out[t] = Y[((size_t)b * NPTS + n) * 64 + o] + bo[o];
}

// ---------------------------- host-side helpers ----------------------------
static inline int rnd16(int x) { return (x + 15) & ~15; }

static void launch_gemm(hipStream_t s, const h16* X, const h16* Wt, float* Y,
                        int M, int Kp, int Copad) {
  if ((M % 32) == 0 && (Copad % 64) == 0) {
    gemm_wmma_2x4<<<dim3((unsigned)(M / 32), (unsigned)(Copad / 64)), 32, 0, s>>>(
        X, Wt, Y, Kp, Copad);
  } else {
    gemm_wmma_1<<<dim3((unsigned)(M / 16), (unsigned)(Copad / 16)), 32, 0, s>>>(
        X, Wt, Y, Kp, Copad);
  }
}

static void run_cbr(hipStream_t s,
                    const float* W, const float* g, const float* b,
                    const h16* Xin, int M, int Ci, int Kp, int Co,
                    h16* Wbuf, float* Yk, float* mean, float* rstd,
                    h16* Xout, int dstStride, int dstOff, int padTo) {
  const int Copad = rnd16(Co);
  {
    long long tot = (long long)Copad * Kp;
    pack_w<<<dim3((unsigned)((tot + 255) / 256)), 256, 0, s>>>(W, Wbuf, Co, Ci, Copad, Kp);
  }
  launch_gemm(s, Xin, Wbuf, Yk, M, Kp, Copad);
  col_stats<<<Co, 256, 0, s>>>(Yk, M, Copad, mean, rstd);
  {
    long long tot = (long long)M * padTo;
    bn_relu_pack<<<dim3((unsigned)((tot + 255) / 256)), 256, 0, s>>>(
        Yk, Copad, Co, g, b, mean, rstd, Xout, dstStride, dstOff, padTo, M);
  }
}

extern "C" void kernel_launch(void* const* d_in, const int* in_sizes, int n_in,
                              void* d_out, int out_size, void* d_ws, size_t ws_size,
                              hipStream_t stream) {
  (void)in_sizes; (void)n_in; (void)out_size; (void)ws_size;
  const float* pts = (const float*)d_in[0];
  const int*   cls = (const int*)d_in[1];
  const float* P[49];
  for (int i = 0; i < 49; ++i) P[i] = (const float*)d_in[2 + i];

  char* base = (char*)d_ws;
  size_t off = 0;
  auto alloc = [&](size_t bytes) -> void* {
    off = (off + 255) & ~(size_t)255;
    void* r = base + off;
    off += bytes;
    return r;
  };

  float* xt0  = (float*)alloc((size_t)BN_PTS * 3 * 4);
  float* xt1  = (float*)alloc((size_t)BN_PTS * 3 * 4);
  float* xcur = (float*)alloc((size_t)BN_PTS * 64 * 4);
  int*   idx  = (int*)  alloc((size_t)M_EDGE * 4);
  h16*   H1   = (h16*)  alloc((size_t)M_EDGE * 128 * 2);
  h16*   H2   = (h16*)  alloc((size_t)M_EDGE * 128 * 2);
  float* Yk   = (float*)alloc((size_t)M_EDGE * 128 * 4);
  h16*   Pp1  = (h16*)  alloc((size_t)BN_PTS * 1280 * 2);
  h16*   Pp2  = (h16*)  alloc((size_t)BN_PTS * 1280 * 2);
  h16*   cat  = (h16*)  alloc((size_t)BN_PTS * 192 * 2);
  h16*   gmx  = (h16*)  alloc((size_t)BD * 1024 * 2);
  h16*   lab  = (h16*)  alloc((size_t)BD * 64 * 2);
  h16*   sA   = (h16*)  alloc((size_t)BD * 1280 * 2);
  h16*   sB   = (h16*)  alloc((size_t)BD * 1280 * 2);
  h16*   Wbuf = (h16*)  alloc((size_t)1024 * 1280 * 2);
  float* mean = (float*)alloc(1024 * 4);
  float* rstd = (float*)alloc(1024 * 4);
  float* trns = (float*)alloc((size_t)BD * 9 * 4);

  // input transpose -> channels-last points
  transpose_in<<<(BN_PTS * 3 + 255) / 256, 256, 0, stream>>>(pts, xt0);

  // ---------------- TNet ----------------
  knn_kernel<<<dim3((NPTS + 63) / 64, BD), 64, 0, stream>>>(xt0, 3, 3, idx);
  build_edge<<<(M_EDGE + 255) / 256, 256, 0, stream>>>(xt0, 3, 3, idx, H1, 32);
  run_cbr(stream, P[0], P[1], P[2],  H1, M_EDGE, 6, 32, 64,  Wbuf, Yk, mean, rstd, H2, 64, 0, 64);
  run_cbr(stream, P[3], P[4], P[5],  H2, M_EDGE, 64, 64, 128, Wbuf, Yk, mean, rstd, H1, 128, 0, 128);
  max_over_k<<<BN_PTS, 64, 0, stream>>>(H1, 128, 128, Pp1, 128, 0, (float*)nullptr, 0);
  run_cbr(stream, P[6], P[7], P[8],  Pp1, BN_PTS, 128, 128, 1024, Wbuf, Yk, mean, rstd, Pp2, 1024, 0, 1024);
  max_over_n<<<dim3(16, BD), 64, 0, stream>>>(Pp2, 1024, 1024, sA, 1024);
  run_cbr(stream, P[9],  P[10], P[11], sA, BD, 1024, 1024, 512, Wbuf, Yk, mean, rstd, sB, 512, 0, 512);
  run_cbr(stream, P[12], P[13], P[14], sB, BD, 512, 512, 256,  Wbuf, Yk, mean, rstd, sA, 256, 0, 256);
  pack_w<<<(16 * 256 + 255) / 256, 256, 0, stream>>>(P[15], Wbuf, 9, 256, 16, 256);
  gemm_wmma_1<<<dim3(1, 1), 32, 0, stream>>>(sA, Wbuf, Yk, 256, 16);
  make_trans_k<<<1, BD * 9, 0, stream>>>(Yk, P[16], trns);
  apply_trans_k<<<(BN_PTS + 255) / 256, 256, 0, stream>>>(xt0, trns, xt1);

  // ---------------- EdgeConv block 1 (3ch input) ----------------
  knn_kernel<<<dim3((NPTS + 63) / 64, BD), 64, 0, stream>>>(xt1, 3, 3, idx);
  build_edge<<<(M_EDGE + 255) / 256, 256, 0, stream>>>(xt1, 3, 3, idx, H1, 32);
  run_cbr(stream, P[17], P[18], P[19], H1, M_EDGE, 6, 32, 64,  Wbuf, Yk, mean, rstd, H2, 64, 0, 64);
  run_cbr(stream, P[20], P[21], P[22], H2, M_EDGE, 64, 64, 64, Wbuf, Yk, mean, rstd, H1, 64, 0, 64);
  max_over_k<<<BN_PTS, 64, 0, stream>>>(H1, 64, 64, cat, 192, 0, xcur, 64);

  // ---------------- EdgeConv block 2 (64ch input) ----------------
  knn_kernel<<<dim3((NPTS + 63) / 64, BD), 64, 0, stream>>>(xcur, 64, 64, idx);
  build_edge<<<(M_EDGE + 255) / 256, 256, 0, stream>>>(xcur, 64, 64, idx, H1, 128);
  run_cbr(stream, P[23], P[24], P[25], H1, M_EDGE, 128, 128, 64, Wbuf, Yk, mean, rstd, H2, 64, 0, 64);
  run_cbr(stream, P[26], P[27], P[28], H2, M_EDGE, 64, 64, 64,  Wbuf, Yk, mean, rstd, H1, 64, 0, 64);
  max_over_k<<<BN_PTS, 64, 0, stream>>>(H1, 64, 64, cat, 192, 64, xcur, 64);

  // ---------------- EdgeConv block 3 (64ch input, one conv) ----------------
  knn_kernel<<<dim3((NPTS + 63) / 64, BD), 64, 0, stream>>>(xcur, 64, 64, idx);
  build_edge<<<(M_EDGE + 255) / 256, 256, 0, stream>>>(xcur, 64, 64, idx, H1, 128);
  run_cbr(stream, P[29], P[30], P[31], H1, M_EDGE, 128, 128, 64, Wbuf, Yk, mean, rstd, H2, 64, 0, 64);
  max_over_k<<<BN_PTS, 64, 0, stream>>>(H2, 64, 64, cat, 192, 128, (float*)nullptr, 0);

  // ---------------- Head ----------------
  run_cbr(stream, P[35], P[36], P[37], cat, BN_PTS, 192, 192, 1024, Wbuf, Yk, mean, rstd, Pp1, 1024, 0, 1024);
  max_over_n<<<dim3(16, BD), 64, 0, stream>>>(Pp1, 1024, 1024, gmx, 1024);
  onehot_k<<<2, 256, 0, stream>>>(cls, sA);
  run_cbr(stream, P[32], P[33], P[34], sA, BD, 16, 32, 64, Wbuf, Yk, mean, rstd, lab, 64, 0, 64);
  {
    long long tot = (long long)BN_PTS * 1280;
    build_xc_k<<<dim3((unsigned)((tot + 255) / 256)), 256, 0, stream>>>(gmx, lab, cat, Pp2);
  }
  run_cbr(stream, P[38], P[39], P[40], Pp2, BN_PTS, 1280, 1280, 256, Wbuf, Yk, mean, rstd, Pp1, 256, 0, 256);
  run_cbr(stream, P[41], P[42], P[43], Pp1, BN_PTS, 256, 256, 256,  Wbuf, Yk, mean, rstd, Pp2, 256, 0, 256);
  run_cbr(stream, P[44], P[45], P[46], Pp2, BN_PTS, 256, 256, 128,  Wbuf, Yk, mean, rstd, Pp1, 128, 0, 128);

  // logit: (32768,128) x (50,128)^T  (Co padded to 64), + bias, transpose out
  pack_w<<<(64 * 128 + 255) / 256, 256, 0, stream>>>(P[47], Wbuf, 50, 128, 64, 128);
  launch_gemm(stream, Pp1, Wbuf, Yk, BN_PTS, 128, 64);
  {
    long long tot = (long long)BD * NSEG * NPTS;
    finalize_k<<<dim3((unsigned)((tot + 255) / 256)), 256, 0, stream>>>(Yk, P[48], (float*)d_out);
  }
}
